// Beran_66649302499984
// MI455X (gfx1250) — matrix operand order repaired and checked by reading, not compiled, not run on hardware
//
#include <hip/hip_runtime.h>
#include <hip/hip_bf16.h>
#include <math.h>

typedef __attribute__((ext_vector_type(16))) _Float16 v16h;
typedef __attribute__((ext_vector_type(8)))  _Float16 v8h;
typedef __attribute__((ext_vector_type(8)))  float    v8f;

#define C_DIM 16
#define B_DIM 256
#define K_DIM 8
#define N_DIM 8192
#define CK    128           // C_DIM * K_DIM (WMMA K dimension, 4 steps of 32)
#define EPSF  1e-13f
#define TOLF  (1e-8f + 1e-5f)   // torch.isclose default vs 1.0

// ---- workspace layout (bytes) ----
static constexpr size_t OFF_P     = 0;                                   // B*CK f16   = 64 KB
static constexpr size_t OFF_H     = OFF_P     + (size_t)B_DIM*CK*2;      // N*CK f16   = 2 MB
static constexpr size_t OFF_SUMSQ = OFF_H     + (size_t)N_DIM*CK*2;      // C*B f32
static constexpr size_t OFF_BASE  = OFF_SUMSQ + (size_t)C_DIM*B_DIM*4;   // B f32
static constexpr size_t OFF_WTS   = ((OFF_BASE + (size_t)B_DIM*4 + 255)/256)*256; // B*N f32 = 8 MB

// ============ kernel 1: softmax over K=8 per (c,b); pack f16 P[b][c*8+k]; sumsq[c*B+b] ============
__global__ __launch_bounds__(256) void k_softmax(const float* __restrict__ cp,
                                                 _Float16* __restrict__ P,
                                                 float* __restrict__ sumsq) {
    int idx = blockIdx.x * 256 + threadIdx.x;          // 0..C*B-1 ; idx = c*B + b
    int c = idx >> 8;
    int b = idx & 255;
    const float* src = cp + (size_t)idx * K_DIM;       // c_p[(c*B+b)*K + k]
    float v[K_DIM];
    float m = src[0];
    #pragma unroll
    for (int j = 0; j < K_DIM; ++j) { v[j] = src[j]; m = fmaxf(m, v[j]); }
    float s = 0.f;
    #pragma unroll
    for (int j = 0; j < K_DIM; ++j) { v[j] = expf(v[j] - m); s += v[j]; }
    float inv = 1.f / s;
    float ss = 0.f;
    v8h h;
    #pragma unroll
    for (int j = 0; j < K_DIM; ++j) {
        float p = v[j] * inv;
        ss += p * p;
        h[j] = (_Float16)p;
    }
    *(v8h*)(P + (size_t)b * CK + c * K_DIM) = h;
    sumsq[idx] = ss;
}

// ============ kernel 2: base[b] = C + sum_c sumsq[c,b] ============
__global__ __launch_bounds__(256) void k_base(const float* __restrict__ sumsq,
                                              float* __restrict__ base) {
    int b = threadIdx.x;
    float acc = (float)C_DIM;
    #pragma unroll
    for (int c = 0; c < C_DIM; ++c) acc += sumsq[c * B_DIM + b];
    base[b] = acc;
}

// ============ kernel 3: one-hot H[n][c*8+k] = (c_in[n,c]==k) in f16 ============
__global__ __launch_bounds__(256) void k_onehot(const int* __restrict__ cin,
                                                _Float16* __restrict__ H) {
    int idx = blockIdx.x * 256 + threadIdx.x;          // 0..N*C-1 ; idx = n*C + c
    int n = idx >> 4;
    int c = idx & 15;
    int lab = cin[idx];
    v8h oh;
    #pragma unroll
    for (int j = 0; j < K_DIM; ++j) oh[j] = (_Float16)(j == lab ? 1.f : 0.f);
    *(v8h*)(H + (size_t)n * CK + c * K_DIM) = oh;
}

// ============ kernel 4: WMMA GEMM dot[b,n] = P·Hᵀ, fused weights = exp(-(base-2dot)/bw) ============
// one wave per (bTile, 8 nTiles); A kept in regs across n-tiles; 4 k-steps of v_wmma_f32_16x16x32_f16
__global__ __launch_bounds__(256) void k_gemm(const _Float16* __restrict__ P,
                                              const _Float16* __restrict__ H,
                                              const float* __restrict__ base,
                                              const float* __restrict__ bwp,
                                              float* __restrict__ wts) {
    const int lane    = threadIdx.x & 31;
    const int wid     = threadIdx.x >> 5;
    const int wg      = blockIdx.x * 8 + wid;   // 0..1023
    const int bTile   = wg >> 6;                // 0..15
    const int nGroup  = wg & 63;                // 0..63
    const int lane15  = lane & 15;
    const int laneSel = lane >> 4;              // 0 | 1

    const float bw     = fminf(fmaxf(bwp[0], 0.1f), 10.0f);
    const float ninvbw = -1.0f / bw;            // hoisted: one rcp instead of 8 divs / tile

    // A operand: 16x32 f16 per ISA layout. lanes 0-15: K {kb..kb+7, kb+16..kb+23};
    // lanes 16-31: K {kb+8..kb+15, kb+24..kb+31}. Row = bTile*16 + lane15.
    const _Float16* Prow = P + (size_t)(bTile * 16 + lane15) * CK;
    v16h A[4];
    #pragma unroll
    for (int ks = 0; ks < 4; ++ks) {
        int k0 = ks * 32 + laneSel * 8;
        v8h lo = *(const v8h*)(Prow + k0);
        v8h hi = *(const v8h*)(Prow + k0 + 16);
        #pragma unroll
        for (int j = 0; j < 8; ++j) { A[ks][j] = lo[j]; A[ks][j + 8] = hi[j]; }
    }
    float baseV[8];
    #pragma unroll
    for (int v = 0; v < 8; ++v) baseV[v] = base[bTile * 16 + laneSel * 8 + v];

    #pragma unroll 1
    for (int nt = 0; nt < 8; ++nt) {
        const int n0 = (nGroup * 8 + nt) * 16;
        // B operand (= Hᵀ tile): column n0+lane15, K {laneSel*8..+7, 16+laneSel*8..+7} (same as A).
        const _Float16* Hrow = H + (size_t)(n0 + lane15) * CK + laneSel * 8;
        v8f acc = {};
        #pragma unroll
        for (int ks = 0; ks < 4; ++ks) {
            v8h lo = *(const v8h*)(Hrow + ks * 32);
            v8h hi = *(const v8h*)(Hrow + ks * 32 + 16);
            v16h Bv;
            #pragma unroll
            for (int j = 0; j < 8; ++j) { Bv[j] = lo[j]; Bv[j + 8] = hi[j]; }
            acc = __builtin_amdgcn_wmma_f32_16x16x32_f16(
                /*neg_a=*/false, A[ks], /*neg_b=*/false, Bv,
                /*c_mod=*/(short)0, acc, /*reuse_a=*/false, /*reuse_b=*/false);
        }
        const int n = n0 + lane15;
        #pragma unroll
        for (int v = 0; v < 8; ++v) {
            int m = bTile * 16 + laneSel * 8 + v;        // C/D layout: M = v + laneSel*8
            float metric = baseV[v] - 2.0f * acc[v];
            wts[(size_t)m * N_DIM + n] = expf(metric * ninvbw);
        }
    }
}

// ============ helpers for row kernel ============
__device__ __forceinline__ float block_sum(float x, float* scr, int tid) {
    scr[tid] = x; __syncthreads();
    #pragma unroll
    for (int off = 128; off > 0; off >>= 1) {
        if (tid < off) scr[tid] += scr[tid + off];
        __syncthreads();
    }
    float s = scr[0];
    __syncthreads();
    return s;
}

__device__ __forceinline__ float block_incl_scan(float x, float* scr, int tid, float& carry) {
    scr[tid] = x; __syncthreads();
    #pragma unroll
    for (int off = 1; off < 256; off <<= 1) {
        float t = (tid >= off) ? scr[tid - off] : 0.f;
        __syncthreads();
        scr[tid] += t;
        __syncthreads();
    }
    float incl  = scr[tid] + carry;
    float total = scr[255];
    __syncthreads();
    carry += total;
    return incl;
}

// ============ kernel 5: per-row Beran pipeline (one block per b) ============
__global__ __launch_bounds__(256) void k_row(const float* __restrict__ wts,
                                             const float* __restrict__ delta,
                                             float* __restrict__ out_sf,
                                             float* __restrict__ out_st) {
    __shared__ float lds_sf[N_DIM];   // 32 KB: surv_func row
    __shared__ float scr[256];
    const int tid = threadIdx.x;
    const int b   = blockIdx.x;
    const float* wrow = wts + (size_t)b * N_DIM;

    // pass A: load row, total weight sum
    float wreg[32];
    float part = 0.f;
    #pragma unroll
    for (int k = 0; k < 32; ++k) { wreg[k] = wrow[k * 256 + tid]; part += wreg[k]; }
    float s = block_sum(part, scr, tid);
    const bool sbad = (s < EPSF);

    // pass B: normalized cumsum -> xi -> hazard cumsum -> surv_func
    float carryW = 0.f, carryH = 0.f;
    for (int k = 0; k < 32; ++k) {
        int i = k * 256 + tid;
        float Wn = sbad ? 0.f : (wreg[k] / s);
        float wc = block_incl_scan(Wn, scr, tid, carryW);
        float shifted = wc - Wn;
        bool bad = (fabsf(shifted - 1.f) <= TOLF) || (fabsf(wc - 1.f) <= TOLF);
        if (bad) { shifted = 0.f; wc = 0.f; }
        float xi   = logf(1.f - shifted) - logf(1.f - wc);
        float filt = delta[i] * xi;
        float haz  = block_incl_scan(filt, scr, tid, carryH);
        float sf   = expf(-haz);
        out_sf[(size_t)b * N_DIM + i] = sf;
        lds_sf[i] = sf;
    }
    __syncthreads();

    // pass C: surv_steps (telescoping) sum, then normalized write
    float part2 = 0.f;
    for (int k = 0; k < 32; ++k) {
        int i = k * 256 + tid;
        float st = (i == 0) ? (1.f - lds_sf[0]) : (lds_sf[i - 1] - lds_sf[i]);
        part2 += st;
    }
    float s2 = block_sum(part2, scr, tid);
    const bool s2bad = (s2 < EPSF);
    for (int k = 0; k < 32; ++k) {
        int i = k * 256 + tid;
        float st = (i == 0) ? (1.f - lds_sf[0]) : (lds_sf[i - 1] - lds_sf[i]);
        out_st[(size_t)b * N_DIM + i] = s2bad ? 0.f : (st / s2);
    }
}

extern "C" void kernel_launch(void* const* d_in, const int* in_sizes, int n_in,
                              void* d_out, int out_size, void* d_ws, size_t ws_size,
                              hipStream_t stream) {
    const float* c_p   = (const float*)d_in[0];   // [C,B,K] f32
    const int*   c_in  = (const int*)d_in[1];     // [N,C] i32
    const float* delta = (const float*)d_in[2];   // [N] f32
    const float* bwp   = (const float*)d_in[3];   // [1] f32
    float* out = (float*)d_out;                   // [2, B, N] f32 (surv_func, surv_steps)

    char* ws = (char*)d_ws;
    _Float16* P     = (_Float16*)(ws + OFF_P);
    _Float16* H     = (_Float16*)(ws + OFF_H);
    float*    sumsq = (float*)(ws + OFF_SUMSQ);
    float*    base  = (float*)(ws + OFF_BASE);
    float*    wts   = (float*)(ws + OFF_WTS);

    k_softmax<<<dim3((C_DIM * B_DIM) / 256), dim3(256), 0, stream>>>(c_p, P, sumsq);
    k_base   <<<dim3(1),                      dim3(256), 0, stream>>>(sumsq, base);
    k_onehot <<<dim3((N_DIM * C_DIM) / 256),  dim3(256), 0, stream>>>(c_in, H);
    k_gemm   <<<dim3(128),                    dim3(256), 0, stream>>>(P, H, base, bwp, wts);
    k_row    <<<dim3(B_DIM),                  dim3(256), 0, stream>>>(wts, delta,
                                                                     out,
                                                                     out + (size_t)B_DIM * N_DIM);
}
